// LongcatFlashAttention_29300266893623
// MI455X (gfx1250) — matrix operand order repaired and checked by reading, not compiled, not run on hardware
//
#include <hip/hip_runtime.h>
#include <hip/hip_bf16.h>

// ---------------------------------------------------------------------------
// MLA (Longcat) attention block for gfx1250 — WMMA f16 path, f32 accumulate,
// async global->LDS staging (GLOBAL_LOAD_ASYNC_TO_LDS_B128 / ASYNCcnt).
// ---------------------------------------------------------------------------

typedef _Float16 half_t;
typedef __attribute__((ext_vector_type(16))) half_t v16h;
typedef __attribute__((ext_vector_type(8)))  half_t v8h;
typedef __attribute__((ext_vector_type(8)))  float  v8f;

// Problem constants
#define S_LEN   1536
#define HID     6144
#define QL      1536
#define NH      64
#define DN      128
#define DR      64
#define DV      128
#define KVL     512
#define DCAT    576           // absorbed (512) + rope (64)
#define QW      12288         // NH*(DN+DR)
#define CTXW    32768         // NH*KVL
#define OVW     8192          // NH*DV

#define SM_SCALE  0.07216878364870322f   // 1/sqrt(192)
#define SCALE_Q   2.0f                   // sqrt(6144/1536)
#define SCALE_KV  3.4641016151377544f    // sqrt(6144/512)
#define EPSF      1e-6f

// ---------------------------------------------------------------------------
// CDNA5 async copy helpers (CDNA5 ISA): global -> LDS without
// touching VGPR data paths; completion tracked with ASYNCcnt.
// LDS address = low 32 bits of the generic shared pointer (flat->LDS
// truncation, ISA §10.2 aperture rules).
// ---------------------------------------------------------------------------
__device__ __forceinline__ unsigned lds_off(const void* p) {
    return (unsigned)(unsigned long long)p;
}

__device__ __forceinline__ void async_load_b128(unsigned ldsaddr, const void* gptr) {
    asm volatile("global_load_async_to_lds_b128 %0, %1, off"
                 :: "v"(ldsaddr), "v"(gptr)
                 : "memory");
}

__device__ __forceinline__ void wait_async0() {
    asm volatile("s_wait_asynccnt 0x0" ::: "memory");
}

// ---------------------------------------------------------------------------
// WMMA fragment loaders (wave32; contiguous k order used by this code for both A and B)
// A 16x32 f16: lane<16 -> row=lane, K {0..15}; lane>=16 -> row=lane-16,
//              K {16..31}
// B 32x16 f16: lane<16 -> col=lane, K {0..15};       lane>=16 -> K {16..31}
// C/D 16x16 f32: elem r -> M = r + 8*(lane>=16), N = lane&15
// ---------------------------------------------------------------------------
__device__ __forceinline__ v16h load_frag_a(const half_t* base, int stride, int lane) {
    const int r  = lane & 15;
    const int kb = (lane < 16) ? 0 : 16;
    v8h a0 = *(const v8h*)(base + r * stride + kb);
    v8h a1 = *(const v8h*)(base + r * stride + kb + 8);
    v16h out;
#pragma unroll
    for (int i = 0; i < 8; ++i) { out[i] = a0[i]; out[8 + i] = a1[i]; }
    return out;
}

__device__ __forceinline__ v16h load_frag_b(const half_t* base, int stride, int lane) {
    const int c  = lane & 15;
    const int kb = (lane < 16) ? 0 : 16;
    v8h b0 = *(const v8h*)(base + c * stride + kb);
    v8h b1 = *(const v8h*)(base + c * stride + kb + 8);
    v16h out;
#pragma unroll
    for (int i = 0; i < 8; ++i) { out[i] = b0[i]; out[8 + i] = b1[i]; }
    return out;
}

__device__ __forceinline__ v8f wmma_f16(v16h a, v16h b, v8f c) {
    return __builtin_amdgcn_wmma_f32_16x16x32_f16(false, a, false, b, (short)0, c,
                                                  false, false);
}

// ---------------------------------------------------------------------------
// f32 -> f16 conversion (grid-stride)
// ---------------------------------------------------------------------------
__global__ void cvt_f32_f16_kernel(const float* __restrict__ in,
                                   half_t* __restrict__ out, long long n) {
    long long i      = (long long)blockIdx.x * blockDim.x + threadIdx.x;
    long long stride = (long long)gridDim.x * blockDim.x;
    for (; i < n; i += stride) out[i] = (half_t)in[i];
}

// ---------------------------------------------------------------------------
// Generic tiled WMMA GEMM: C[M,N] = A[M,K] @ B[K,N]
//   A,B f16 row-major, C f16 or f32 row-major. 256 threads = 8 waves.
//   Block tile 128x64; wave w owns rows 16w..16w+15, all 64 cols.
//   grid = (N/64, M/128, batch). M%128==0, N%64==0, K%32==0.
//   A tile staged with async global->LDS; B tile staged transposed via VGPRs.
// ---------------------------------------------------------------------------
template <bool OUT_HALF>
__global__ __launch_bounds__(256)
void gemm_wmma_kernel(const half_t* __restrict__ A, const half_t* __restrict__ B,
                      void* __restrict__ Cv, int K, int lda, int ldb, int ldc,
                      long long sA, long long sB, long long sC) {
    __shared__ half_t As[128 * 32];     // [m][k]      8 KB
    __shared__ half_t Bs[64 * 32];      // [n][k] (T)  4 KB

    const int tid  = threadIdx.x;
    const int lane = tid & 31;
    const int wave = tid >> 5;
    const long long m0 = (long long)blockIdx.y * 128;
    const long long n0 = (long long)blockIdx.x * 64;

    A += (long long)blockIdx.z * sA;
    B += (long long)blockIdx.z * sB;
    half_t* Ch = (half_t*)Cv;
    float*  Cf = (float*)Cv;
    const long long cbase = (long long)blockIdx.z * sC;

    v8f acc[4];
#pragma unroll
    for (int t = 0; t < 4; ++t)
#pragma unroll
        for (int r = 0; r < 8; ++r) acc[t][r] = 0.0f;

    const int arow = tid >> 1;           // 0..127
    const int aseg = (tid & 1) * 16;     // 0 / 16 (k offset, halves)
    const int brow = tid >> 3;           // 0..31 (k)
    const int bseg = (tid & 7) * 8;      // 0..56 (n offset, halves)

    for (int k0 = 0; k0 < K; k0 += 32) {
        // ---- stage A tile: async global -> LDS (2 x b128 per thread) ----
        {
            const half_t* g = A + (m0 + arow) * lda + k0 + aseg;
            async_load_b128(lds_off(As + arow * 32 + aseg),     g);
            async_load_b128(lds_off(As + arow * 32 + aseg + 8), g + 8);
        }
        // ---- stage B tile transposed ([n][k]) via VGPRs ----
        {
            const v8h bv = *(const v8h*)(B + (long long)(k0 + brow) * ldb + n0 + bseg);
#pragma unroll
            for (int i = 0; i < 8; ++i) Bs[(bseg + i) * 32 + brow] = bv[i];
            if (k0 + 32 < K)
                __builtin_prefetch(B + (long long)(k0 + 32 + brow) * ldb + n0 + bseg,
                                   0, 3);
        }
        wait_async0();
        __syncthreads();

        v16h a = load_frag_a(As + wave * 16 * 32, 32, lane);
#pragma unroll
        for (int t = 0; t < 4; ++t) {
            v16h b = load_frag_b(Bs + t * 16 * 32, 32, lane);
            acc[t] = wmma_f16(a, b, acc[t]);
        }
        __syncthreads();
    }

    // ---- epilogue ----
    const int nloc = lane & 15;
    const int moff = (lane >> 4) * 8;
#pragma unroll
    for (int t = 0; t < 4; ++t) {
#pragma unroll
        for (int r = 0; r < 8; ++r) {
            long long m = m0 + wave * 16 + moff + r;
            long long n = n0 + t * 16 + nloc;
            if (OUT_HALF)
                Ch[cbase + m * ldc + n] = (half_t)acc[t][r];
            else
                Cf[cbase + m * ldc + n] = acc[t][r];
        }
    }
}

// ---------------------------------------------------------------------------
// RMSNorm (+scale) : f32 in -> f16 out, one block per row
// ---------------------------------------------------------------------------
__global__ __launch_bounds__(256)
void rmsnorm_kernel(const float* __restrict__ in, const float* __restrict__ w,
                    half_t* __restrict__ out, int width, int istride, int ostride,
                    float scale) {
    __shared__ float red[256];
    const int row = blockIdx.x;
    const float* x = in + (long long)row * istride;
    float s = 0.0f;
    for (int c = threadIdx.x; c < width; c += 256) {
        float v = x[c];
        s += v * v;
    }
    red[threadIdx.x] = s;
    __syncthreads();
    for (int o = 128; o > 0; o >>= 1) {
        if (threadIdx.x < o) red[threadIdx.x] += red[threadIdx.x + o];
        __syncthreads();
    }
    const float r = rsqrtf(red[0] / (float)width + EPSF) * scale;
    for (int c = threadIdx.x; c < width; c += 256)
        out[(long long)row * ostride + c] = (half_t)(w[c] * x[c] * r);
}

// ---------------------------------------------------------------------------
// RoPE for k_pe: latent f32 cols [512,576) -> kcat f16 cols [512,576)
// ---------------------------------------------------------------------------
__global__ void rope_k_kernel(const float* __restrict__ latent,
                              const float* __restrict__ cosb,
                              const float* __restrict__ sinb,
                              half_t* __restrict__ kcat) {
    int idx = blockIdx.x * 256 + threadIdx.x;
    if (idx >= S_LEN * DR) return;
    int s = idx >> 6, i = idx & 63;
    float x = latent[(long long)s * DCAT + KVL + i];
    float o = (i < 32) ? -latent[(long long)s * DCAT + KVL + i + 32]
                       :  latent[(long long)s * DCAT + KVL + i - 32];
    kcat[(long long)s * DCAT + KVL + i] =
        (half_t)(x * cosb[s * DR + i] + o * sinb[s * DR + i]);
}

// ---------------------------------------------------------------------------
// RoPE for q_pe: q f16 [s, h*192+128 .. +64) -> qcat f16 [s, h, 512..576)
// ---------------------------------------------------------------------------
__global__ void rope_q_kernel(const half_t* __restrict__ q,
                              const float* __restrict__ cosb,
                              const float* __restrict__ sinb,
                              half_t* __restrict__ qcat) {
    long long idx = (long long)blockIdx.x * 256 + threadIdx.x;
    if (idx >= (long long)S_LEN * NH * DR) return;
    int i = (int)(idx & 63);
    long long sh = idx >> 6;
    int h = (int)(sh & 63);
    long long s = sh >> 6;
    long long base = s * QW + (long long)h * (DN + DR) + DN;
    float x = (float)q[base + i];
    float o = (i < 32) ? -(float)q[base + i + 32] : (float)q[base + i - 32];
    qcat[s * (NH * DCAT) + (long long)h * DCAT + KVL + i] =
        (half_t)(x * cosb[s * DR + i] + o * sinb[s * DR + i]);
}

// ---------------------------------------------------------------------------
// Flash attention over qcat (s,h,576) x kcat (t,576); ctx (s,h,512) f16.
// Block = (32 q-rows, one head), 256 threads = 8 waves.
// Wave layout: rt = wave&1 (16 q rows), cg = wave>>1 (128 ctx cols).
// Q/K tiles staged with async global->LDS; V tile (=K tile cols 0..511)
// transposed LDS->LDS — no second pass over HBM.
// ---------------------------------------------------------------------------
struct __align__(16) AttSmem {
    half_t Qs[32 * DCAT];    // 36864 B
    half_t Ks[32 * DCAT];    // 36864 B
    half_t Vt[KVL * 32];     // 32768 B, transposed [col][key]
    float  Ss[32 * 32];      //  4096 B
    half_t Ps[32 * 32];      //  2048 B
    float  srow[32];         //   128 B
};

__global__ __launch_bounds__(256)
void attention_kernel(const half_t* __restrict__ qcat,
                      const half_t* __restrict__ kcat,
                      half_t* __restrict__ ctx) {
    extern __shared__ char smem_raw[];
    AttSmem& sm = *(AttSmem*)smem_raw;

    const int tid  = threadIdx.x;
    const int lane = tid & 31;
    const int wave = tid >> 5;
    const int h    = blockIdx.y;
    const int qb   = blockIdx.x * 32;
    const int rt   = wave & 1;     // q row tile (16 rows)
    const int cg   = wave >> 1;    // ctx col group (128 cols)

    // stage Q tile once (async): 32 x 576 halves
    for (int c = tid; c < 32 * (DCAT / 8); c += 256) {
        int row = c / (DCAT / 8), col = (c % (DCAT / 8)) * 8;
        async_load_b128(lds_off(sm.Qs + row * DCAT + col),
                        qcat + (long long)(qb + row) * (NH * DCAT) +
                            (long long)h * DCAT + col);
    }

    float m_run = -3.0e38f, l_run = 0.0f;   // per-row state (threads 0..31)
    v8f acc[8];
#pragma unroll
    for (int t = 0; t < 8; ++t)
#pragma unroll
        for (int r = 0; r < 8; ++r) acc[t][r] = 0.0f;

    const int nkt = qb / 32 + 1;            // causal frontier
    for (int j = 0; j < nkt; ++j) {
        const int kb = j * 32;
        __syncthreads();   // previous P@V reads done before restaging

        // stage K tile (row-major [key][576]) async
        for (int c = tid; c < 32 * (DCAT / 8); c += 256) {
            int row = c / (DCAT / 8), col = (c % (DCAT / 8)) * 8;
            async_load_b128(lds_off(sm.Ks + row * DCAT + col),
                            kcat + (long long)(kb + row) * DCAT + col);
        }
        wait_async0();        // also covers the Q-tile loads on iter 0
        __syncthreads();

        // build transposed V tile [col][key] from staged K tile (LDS->LDS)
        for (int c = tid; c < 2048; c += 256) {
            int key = c >> 6, cb = (c & 63) * 8;
            v8h v = *(const v8h*)(sm.Ks + key * DCAT + cb);
#pragma unroll
            for (int i = 0; i < 8; ++i) sm.Vt[(cb + i) * 32 + key] = v[i];
        }
        __syncthreads();

        // ---- scores: waves 0..3 compute 16x16 tiles of the 32x32 score block
        if (wave < 4) {
            const int srt = wave & 1, skt = wave >> 1;
            v8f sacc;
#pragma unroll
            for (int r = 0; r < 8; ++r) sacc[r] = 0.0f;
            for (int dc = 0; dc < DCAT; dc += 32) {
                v16h a = load_frag_a(sm.Qs + srt * 16 * DCAT + dc, DCAT, lane);
                v16h b = load_frag_b(sm.Ks + skt * 16 * DCAT + dc, DCAT, lane);
                sacc = wmma_f16(a, b, sacc);
            }
            const int n = lane & 15, mo = (lane >> 4) * 8;
#pragma unroll
            for (int r = 0; r < 8; ++r)
                sm.Ss[(srt * 16 + mo + r) * 32 + skt * 16 + n] = sacc[r] * SM_SCALE;
        }
        __syncthreads();

        // ---- online softmax: one thread per q-row
        if (tid < 32) {
            const int row = tid, qg = qb + row;
            float sv[32];
            float mx = m_run;
#pragma unroll
            for (int kk = 0; kk < 32; ++kk) {
                float v = sm.Ss[row * 32 + kk];
                if (kb + kk > qg) v = -3.0e38f;
                sv[kk] = v;
                mx = fmaxf(mx, v);
            }
            const float scale = __expf(m_run - mx);
            float sum = 0.0f;
#pragma unroll
            for (int kk = 0; kk < 32; ++kk) {
                float p = __expf(sv[kk] - mx);
                sum += p;
                sm.Ps[row * 32 + kk] = (half_t)p;
            }
            l_run = l_run * scale + sum;
            m_run = mx;
            sm.srow[row] = scale;
        }
        __syncthreads();

        // ---- rescale + P@V (all 8 waves)
        const int mo = (lane >> 4) * 8;
        float sc[8];
#pragma unroll
        for (int r = 0; r < 8; ++r) sc[r] = sm.srow[rt * 16 + mo + r];
        v16h a = load_frag_a(sm.Ps + rt * 16 * 32, 32, lane);
#pragma unroll
        for (int t = 0; t < 8; ++t) {
#pragma unroll
            for (int r = 0; r < 8; ++r) acc[t][r] *= sc[r];
            v16h b = load_frag_b(sm.Vt + (cg * 128 + t * 16) * 32, 32, lane);
            acc[t] = wmma_f16(a, b, acc[t]);
        }
    }

    __syncthreads();
    if (tid < 32) sm.srow[tid] = 1.0f / l_run;
    __syncthreads();

    {
        const int n = lane & 15, mo = (lane >> 4) * 8;
        float inv[8];
#pragma unroll
        for (int r = 0; r < 8; ++r) inv[r] = sm.srow[rt * 16 + mo + r];
#pragma unroll
        for (int t = 0; t < 8; ++t)
#pragma unroll
            for (int r = 0; r < 8; ++r) {
                long long addr = (long long)(qb + rt * 16 + mo + r) * CTXW +
                                 (long long)h * KVL + cg * 128 + t * 16 + n;
                ctx[addr] = (half_t)(acc[t][r] * inv[r]);
            }
    }
}

// ---------------------------------------------------------------------------
// Host-side orchestration
// ---------------------------------------------------------------------------
extern "C" void kernel_launch(void* const* d_in, const int* in_sizes, int n_in,
                              void* d_out, int out_size, void* d_ws, size_t ws_size,
                              hipStream_t stream) {
    (void)in_sizes; (void)n_in; (void)out_size; (void)ws_size;

    const float* hidden  = (const float*)d_in[0];
    const float* cosb    = (const float*)d_in[1];
    const float* sinb    = (const float*)d_in[2];
    const float* w_qa    = (const float*)d_in[3];
    const float* qa_ln   = (const float*)d_in[4];
    const float* w_qb    = (const float*)d_in[5];
    const float* w_kva   = (const float*)d_in[6];
    const float* kva_ln  = (const float*)d_in[7];
    const float* w_k     = (const float*)d_in[8];
    const float* w_v     = (const float*)d_in[9];
    const float* w_o     = (const float*)d_in[10];
    float* out = (float*)d_out;

    // ---- workspace carve-up (256B aligned) ----
    char* base = (char*)d_ws;
    size_t off = 0;
    auto alloc = [&](size_t bytes) -> char* {
        char* p = base + off;
        off += (bytes + 255) & ~(size_t)255;
        return p;
    };
    half_t* hid_h   = (half_t*)alloc((size_t)S_LEN * HID * 2);        // 18.9 MB
    half_t* wqa_h   = (half_t*)alloc((size_t)HID * QL * 2);           // 18.9 MB
    half_t* wqb_h   = (half_t*)alloc((size_t)QL * QW * 2);            // 37.7 MB
    half_t* wkva_h  = (half_t*)alloc((size_t)HID * DCAT * 2);         //  7.1 MB
    half_t* wk_h    = (half_t*)alloc((size_t)NH * DN * KVL * 2);      //  8.4 MB
    half_t* wv_h    = (half_t*)alloc((size_t)NH * KVL * DV * 2);      //  8.4 MB
    float*  qlat_f  = (float*) alloc((size_t)S_LEN * QL * 4);         //  9.4 MB
    half_t* qlat_h  = (half_t*)alloc((size_t)S_LEN * QL * 2);         //  4.7 MB
    half_t* q_h     = (half_t*)alloc((size_t)S_LEN * QW * 2);         // 37.7 MB (reused as outv)
    float*  lat_f   = (float*) alloc((size_t)S_LEN * DCAT * 4);       //  3.5 MB
    half_t* kcat_h  = (half_t*)alloc((size_t)S_LEN * DCAT * 2);       //  1.8 MB
    half_t* qcat_h  = (half_t*)alloc((size_t)S_LEN * NH * DCAT * 2);  // 113.2 MB (reused as wo_h)
    half_t* ctx_h   = (half_t*)alloc((size_t)S_LEN * CTXW * 2);       // 100.7 MB
    half_t* outv_h  = q_h;      // alias: q free after RoPE + absorb GEMM
    half_t* wo_h    = qcat_h;   // alias: qcat free after attention

    auto cvt = [&](const float* src, half_t* dst, long long n) {
        int blocks = (int)((n + 255) / 256);
        cvt_f32_f16_kernel<<<blocks, 256, 0, stream>>>(src, dst, n);
    };

    // ---- 1. convert activations + early weights to f16 ----
    cvt(hidden, hid_h,  (long long)S_LEN * HID);
    cvt(w_qa,   wqa_h,  (long long)HID * QL);
    cvt(w_qb,   wqb_h,  (long long)QL * QW);
    cvt(w_kva,  wkva_h, (long long)HID * DCAT);
    cvt(w_k,    wk_h,   (long long)NH * DN * KVL);
    cvt(w_v,    wv_h,   (long long)NH * KVL * DV);

    dim3 gblk(256);

    // ---- 2. q_lat = hidden @ w_qa  (f32 out, for RMS norm) ----
    gemm_wmma_kernel<false><<<dim3(QL / 64, S_LEN / 128, 1), gblk, 0, stream>>>(
        hid_h, wqa_h, qlat_f, HID, HID, QL, QL, 0, 0, 0);

    // ---- 3. RMS norm -> qlat_h (f16, *SCALE_Q) ----
    rmsnorm_kernel<<<S_LEN, 256, 0, stream>>>(qlat_f, qa_ln, qlat_h, QL, QL, QL,
                                              SCALE_Q);

    // ---- 4. q = qlat_h @ w_qb  (f16 out) ----
    gemm_wmma_kernel<true><<<dim3(QW / 64, S_LEN / 128, 1), gblk, 0, stream>>>(
        qlat_h, wqb_h, q_h, QL, QL, QW, QW, 0, 0, 0);

    // ---- 5. latent = hidden @ w_kva  (f32 out) ----
    gemm_wmma_kernel<false><<<dim3(DCAT / 64, S_LEN / 128, 1), gblk, 0, stream>>>(
        hid_h, wkva_h, lat_f, HID, HID, DCAT, DCAT, 0, 0, 0);

    // ---- 6. c_kv = rmsnorm(latent[:,:512]) * SCALE_KV -> kcat[:, :512] ----
    rmsnorm_kernel<<<S_LEN, 256, 0, stream>>>(lat_f, kva_ln, kcat_h, KVL, DCAT,
                                              DCAT, SCALE_KV);

    // ---- 7. RoPE ----
    rope_k_kernel<<<(S_LEN * DR + 255) / 256, 256, 0, stream>>>(lat_f, cosb, sinb,
                                                                kcat_h);
    rope_q_kernel<<<(int)(((long long)S_LEN * NH * DR + 255) / 256), 256, 0,
                    stream>>>(q_h, cosb, sinb, qcat_h);

    // ---- 8. absorbed K: qcat[:,h,:512] = q_nope[:,h] @ w_k[h]  (batched) ----
    gemm_wmma_kernel<true><<<dim3(KVL / 64, S_LEN / 128, NH), gblk, 0, stream>>>(
        q_h, wk_h, qcat_h, DN, QW, KVL, NH * DCAT,
        (long long)(DN + DR), (long long)DN * KVL, (long long)DCAT);

    // ---- 9. flash attention -> ctx (s,h,512) ----
    attention_kernel<<<dim3(S_LEN / 32, NH), 256, sizeof(AttSmem), stream>>>(
        qcat_h, kcat_h, ctx_h);

    // ---- 10. out_v[:,h*128..] = ctx[:,h] @ w_v[h]  (batched) ----
    gemm_wmma_kernel<true><<<dim3(DV / 64, S_LEN / 128, NH), gblk, 0, stream>>>(
        ctx_h, wv_h, outv_h, KVL, CTXW, DV, OVW,
        (long long)KVL, (long long)KVL * DV, (long long)DV);

    // ---- 11. convert w_o (into freed qcat region), final projection ----
    cvt(w_o, wo_h, (long long)OVW * HID);
    gemm_wmma_kernel<false><<<dim3(HID / 64, S_LEN / 128, 1), gblk, 0, stream>>>(
        outv_h, wo_h, out, OVW, OVW, HID, HID, 0, 0, 0);
}
